// GatedGCNLayer_1168231104592
// MI455X (gfx1250) — matrix operand order repaired and checked by reading, not compiled, not run on hardware
//
#include <hip/hip_runtime.h>

#define N_NODES 50000
#define N_EDGES 600000
#define DIM 128
#define EPS_AGG 1e-6f
#define EPS_BN  1e-5f

typedef unsigned short u16;
typedef __attribute__((ext_vector_type(16))) __bf16 v16bf;
typedef __attribute__((ext_vector_type(8)))  float  v8f;

struct B32Frag { uint4 lo, hi; };

static __device__ __forceinline__ v16bf make_frag(uint4 lo, uint4 hi) {
    B32Frag t; t.lo = lo; t.hi = hi;
    return __builtin_bit_cast(v16bf, t);
}

// ---------------------------------------------------------------- fp32 -> bf16 (RNE)
__global__ void cvt_bf16_kernel(const float* __restrict__ x, u16* __restrict__ y, int n) {
    int i = blockIdx.x * blockDim.x + threadIdx.x;
    if (i < n) {
        unsigned int u = __builtin_bit_cast(unsigned int, x[i]);
        unsigned int lsb = (u >> 16) & 1u;
        u += 0x7fffu + lsb;
        y[i] = (u16)(u >> 16);
    }
}

// ---------------------------------------------------------------- node GEMMs: Ah,Bh,Dh,Eh
// one workgroup = one 16-row tile of h; 8 waves, wave w owns columns [16w,16w+16) of all 4 outputs
__global__ void __launch_bounds__(256)
node_gemm_kernel(const u16* __restrict__ hbf, const u16* __restrict__ Wbf,
                 const float* __restrict__ Ab, const float* __restrict__ Bb,
                 const float* __restrict__ Db, const float* __restrict__ Eb,
                 float* __restrict__ nodeOut /* [4][N_NODES*DIM] order A,B,D,E */) {
    __shared__ u16 tile[16 * DIM];

    int mt = blockIdx.x;
    const uint4* gsrc = (const uint4*)(hbf + (size_t)mt * 16 * DIM);
    ((uint4*)tile)[threadIdx.x] = gsrc[threadIdx.x];
    __syncthreads();

    int wave = threadIdx.x >> 5;      // == ntile (0..7)
    int lane = threadIdx.x & 31;
    int half = lane >> 4;
    int m    = lane & 15;
    int col  = wave * 16 + m;

    v8f acc[4];
#pragma unroll
    for (int j = 0; j < 4; ++j) {
        const float* bp = (j == 0) ? Ab : (j == 1) ? Bb : (j == 2) ? Db : Eb;
        float bv = bp[col];
#pragma unroll
        for (int r = 0; r < 8; ++r) acc[j][r] = bv;
    }

#pragma unroll
    for (int k0 = 0; k0 < DIM; k0 += 32) {
        // A fragment (16x32 bf16): two 16B chunks per lane from LDS
        const uint4* pa0 = (const uint4*)(tile + m * DIM + k0 + half * 8);
        const uint4* pa1 = (const uint4*)(tile + m * DIM + k0 + 16 + half * 8);
        v16bf a = make_frag(pa0[0], pa1[0]);
#pragma unroll
        for (int j = 0; j < 4; ++j) {
            int slot = (j < 2) ? j : j + 1;   // weight slots: 0=A,1=B,2=C,3=D,4=E
            const u16* Wrow = Wbf + (size_t)slot * DIM * DIM + (size_t)col * DIM + k0;
            const uint4* pb = (const uint4*)Wrow;
            v16bf b = make_frag(pb[half], pb[2 + half]);
            acc[j] = __builtin_amdgcn_wmma_f32_16x16x32_bf16(
                false, a, false, b, (short)0, acc[j], false, false);
        }
    }

#pragma unroll
    for (int j = 0; j < 4; ++j) {
        float* out = nodeOut + (size_t)j * N_NODES * DIM;
#pragma unroll
        for (int r = 0; r < 8; ++r) {
            int row = half * 8 + r;
            out[((size_t)mt * 16 + row) * DIM + col] = acc[j][r];
        }
    }
}

// ---------------------------------------------------------------- edge GEMM + gather + sigmoid + scatter + e-BN stats
__global__ void __launch_bounds__(256)
edge_gemm_kernel(const u16* __restrict__ ebf, const u16* __restrict__ Wbf,
                 const float* __restrict__ Cb,
                 const int* __restrict__ src, const int* __restrict__ dst,
                 const float* __restrict__ Bh, const float* __restrict__ Dh,
                 const float* __restrict__ Eh,
                 float* __restrict__ enew,
                 float* __restrict__ sum_sh, float* __restrict__ sum_s,
                 float* __restrict__ esum, float* __restrict__ esumsq) {
    __shared__ u16 tile[16 * DIM];
    __shared__ int sS[16], sD[16];

    int et = blockIdx.x;
    const uint4* gsrc = (const uint4*)(ebf + (size_t)et * 16 * DIM);
    ((uint4*)tile)[threadIdx.x] = gsrc[threadIdx.x];
    if (threadIdx.x < 16) {
        sS[threadIdx.x] = src[et * 16 + threadIdx.x];
        sD[threadIdx.x] = dst[et * 16 + threadIdx.x];
    }
    __syncthreads();

    int wave = threadIdx.x >> 5;
    int lane = threadIdx.x & 31;
    int half = lane >> 4;
    int m    = lane & 15;
    int col  = wave * 16 + m;

    v8f acc;
    float bv = Cb[col];
#pragma unroll
    for (int r = 0; r < 8; ++r) acc[r] = bv;

    const u16* W = Wbf + (size_t)2 * DIM * DIM;   // slot 2 = C_w
#pragma unroll
    for (int k0 = 0; k0 < DIM; k0 += 32) {
        const uint4* pa0 = (const uint4*)(tile + m * DIM + k0 + half * 8);
        const uint4* pa1 = (const uint4*)(tile + m * DIM + k0 + 16 + half * 8);
        v16bf a = make_frag(pa0[0], pa1[0]);
        const uint4* pb = (const uint4*)(W + (size_t)col * DIM + k0);
        v16bf b = make_frag(pb[half], pb[2 + half]);
        acc = __builtin_amdgcn_wmma_f32_16x16x32_bf16(
            false, a, false, b, (short)0, acc, false, false);
    }

    float cs = 0.f, cs2 = 0.f;
#pragma unroll
    for (int r = 0; r < 8; ++r) {
        int row = half * 8 + r;
        int s = sS[row], d = sD[row];
        float v = acc[r] + Dh[(size_t)s * DIM + col] + Eh[(size_t)d * DIM + col];
        enew[((size_t)et * 16 + row) * DIM + col] = v;
        float sg = 1.0f / (1.0f + __expf(-v));
        atomicAdd(&sum_s[(size_t)d * DIM + col], sg);
        atomicAdd(&sum_sh[(size_t)d * DIM + col], Bh[(size_t)s * DIM + col] * sg);
        cs += v; cs2 += v * v;
    }
    // lane l and l+16 cover the same column -> combine then one atomic per column
    cs  += __shfl_xor(cs, 16, 32);
    cs2 += __shfl_xor(cs2, 16, 32);
    if (half == 0) {
        atomicAdd(&esum[col], cs);
        atomicAdd(&esumsq[col], cs2);
    }
}

// ---------------------------------------------------------------- h_new pre-BN + h column stats
#define ROWS_PER_BLK 256
__global__ void __launch_bounds__(128)
node_combine_kernel(const float* __restrict__ Ah, const float* __restrict__ ssh,
                    const float* __restrict__ ss, float* __restrict__ hpre,
                    float* __restrict__ hsum, float* __restrict__ hsumsq) {
    int c  = threadIdx.x;
    int r0 = blockIdx.x * ROWS_PER_BLK;
    int r1 = r0 + ROWS_PER_BLK; if (r1 > N_NODES) r1 = N_NODES;
    float s = 0.f, s2 = 0.f;
    for (int r = r0; r < r1; ++r) {
        size_t i = (size_t)r * DIM + c;
        float v = Ah[i] + ssh[i] / (ss[i] + EPS_AGG);
        hpre[i] = v;
        s += v; s2 += v * v;
    }
    atomicAdd(&hsum[c], s);
    atomicAdd(&hsumsq[c], s2);
}

// ---------------------------------------------------------------- BN stats finalize
__global__ void finalize_stats_kernel(float* __restrict__ stats) {
    // layout: [hsum, hsumsq, esum, esumsq, mu_h, rs_h, mu_e, rs_e] x 128 floats
    int t = threadIdx.x;
    if (t < 128) {
        float mu  = stats[t] / (float)N_NODES;
        float var = stats[128 + t] / (float)N_NODES - mu * mu;
        stats[512 + t] = mu;
        stats[640 + t] = rsqrtf(var + EPS_BN);
    } else {
        int c = t - 128;
        float mu  = stats[256 + c] / (float)N_EDGES;
        float var = stats[384 + c] / (float)N_EDGES - mu * mu;
        stats[768 + c] = mu;
        stats[896 + c] = rsqrtf(var + EPS_BN);
    }
}

// ---------------------------------------------------------------- BN + ReLU + residual epilogues
__global__ void __launch_bounds__(256)
final_h_kernel(const float* __restrict__ h, const float* __restrict__ hpre,
               const float* __restrict__ stats,
               const float* __restrict__ g, const float* __restrict__ b,
               float* __restrict__ hout) {
    size_t i = (size_t)blockIdx.x * blockDim.x + threadIdx.x;
    int c = (int)(i & (DIM - 1));
    float v = g[c] * (hpre[i] - stats[512 + c]) * stats[640 + c] + b[c];
    hout[i] = h[i] + fmaxf(v, 0.f);
}

__global__ void __launch_bounds__(256)
final_e_kernel(const float* __restrict__ e, const float* __restrict__ stats,
               const float* __restrict__ g, const float* __restrict__ b,
               float* __restrict__ eio /* e_new in, e_out out (in place) */) {
    size_t i = (size_t)blockIdx.x * blockDim.x + threadIdx.x;
    int c = (int)(i & (DIM - 1));
    float v = g[c] * (eio[i] - stats[768 + c]) * stats[896 + c] + b[c];
    eio[i] = e[i] + fmaxf(v, 0.f);
}

// ================================================================ launch
extern "C" void kernel_launch(void* const* d_in, const int* in_sizes, int n_in,
                              void* d_out, int out_size, void* d_ws, size_t ws_size,
                              hipStream_t stream) {
    (void)in_sizes; (void)n_in; (void)out_size; (void)ws_size;

    const float* h   = (const float*)d_in[0];
    const float* e   = (const float*)d_in[1];
    const int*   src = (const int*)d_in[2];
    const int*   dst = (const int*)d_in[3];
    const float* A_w = (const float*)d_in[4];  const float* A_b = (const float*)d_in[5];
    const float* B_w = (const float*)d_in[6];  const float* B_b = (const float*)d_in[7];
    const float* C_w = (const float*)d_in[8];  const float* C_b = (const float*)d_in[9];
    const float* D_w = (const float*)d_in[10]; const float* D_b = (const float*)d_in[11];
    const float* E_w = (const float*)d_in[12]; const float* E_b = (const float*)d_in[13];
    const float* bn_h_g = (const float*)d_in[14]; const float* bn_h_b = (const float*)d_in[15];
    const float* bn_e_g = (const float*)d_in[16]; const float* bn_e_b = (const float*)d_in[17];

    float* h_out = (float*)d_out;
    float* e_new = (float*)d_out + (size_t)N_NODES * DIM;   // also final e_out (in place)

    // workspace carve-up (256B aligned)
    char* p = (char*)d_ws;
    auto take = [&](size_t bytes) -> char* {
        char* q = p;
        p += (bytes + 255) & ~(size_t)255;
        return q;
    };
    u16*   Wbf     = (u16*)take((size_t)5 * DIM * DIM * sizeof(u16));
    u16*   hbf     = (u16*)take((size_t)N_NODES * DIM * sizeof(u16));
    u16*   ebf     = (u16*)take((size_t)N_EDGES * DIM * sizeof(u16));
    float* nodeOut = (float*)take((size_t)4 * N_NODES * DIM * sizeof(float)); // Ah,Bh,Dh,Eh
    float* sum_sh  = (float*)take((size_t)N_NODES * DIM * sizeof(float));
    float* sum_s   = (float*)take((size_t)N_NODES * DIM * sizeof(float));
    float* hpre    = (float*)take((size_t)N_NODES * DIM * sizeof(float));
    float* stats   = (float*)take((size_t)8 * 128 * sizeof(float));

    // zero accumulators (capture-safe async memset)
    hipMemsetAsync(sum_sh, 0, (size_t)N_NODES * DIM * sizeof(float), stream);
    hipMemsetAsync(sum_s,  0, (size_t)N_NODES * DIM * sizeof(float), stream);
    hipMemsetAsync(stats,  0, (size_t)4 * 128 * sizeof(float), stream);

    // bf16 conversions: weight slots 0=A,1=B,2=C,3=D,4=E
    const int WN = DIM * DIM;
    cvt_bf16_kernel<<<(WN + 255) / 256, 256, 0, stream>>>(A_w, Wbf + 0 * (size_t)WN, WN);
    cvt_bf16_kernel<<<(WN + 255) / 256, 256, 0, stream>>>(B_w, Wbf + 1 * (size_t)WN, WN);
    cvt_bf16_kernel<<<(WN + 255) / 256, 256, 0, stream>>>(C_w, Wbf + 2 * (size_t)WN, WN);
    cvt_bf16_kernel<<<(WN + 255) / 256, 256, 0, stream>>>(D_w, Wbf + 3 * (size_t)WN, WN);
    cvt_bf16_kernel<<<(WN + 255) / 256, 256, 0, stream>>>(E_w, Wbf + 4 * (size_t)WN, WN);
    cvt_bf16_kernel<<<(N_NODES * DIM + 255) / 256, 256, 0, stream>>>(h, hbf, N_NODES * DIM);
    cvt_bf16_kernel<<<(N_EDGES * DIM + 255) / 256, 256, 0, stream>>>(e, ebf, N_EDGES * DIM);

    // node GEMMs (Ah, Bh, Dh, Eh)
    node_gemm_kernel<<<N_NODES / 16, 256, 0, stream>>>(
        hbf, Wbf, A_b, B_b, D_b, E_b, nodeOut);

    // edge GEMM + fused gather/sigmoid/scatter + e BN stats
    edge_gemm_kernel<<<N_EDGES / 16, 256, 0, stream>>>(
        ebf, Wbf, C_b, src, dst,
        nodeOut + (size_t)1 * N_NODES * DIM,   // Bh
        nodeOut + (size_t)2 * N_NODES * DIM,   // Dh
        nodeOut + (size_t)3 * N_NODES * DIM,   // Eh
        e_new, sum_sh, sum_s,
        stats + 256, stats + 384);

    // h_new pre-BN + h BN stats
    node_combine_kernel<<<(N_NODES + ROWS_PER_BLK - 1) / ROWS_PER_BLK, 128, 0, stream>>>(
        nodeOut /* Ah */, sum_sh, sum_s, hpre, stats, stats + 128);

    finalize_stats_kernel<<<1, 256, 0, stream>>>(stats);

    final_h_kernel<<<(N_NODES * DIM) / 256, 256, 0, stream>>>(
        h, hpre, stats, bn_h_g, bn_h_b, h_out);
    final_e_kernel<<<(N_EDGES * DIM) / 256, 256, 0, stream>>>(
        e, stats, bn_e_g, bn_e_b, e_new);
}
